// DualTransformerBlock_38276748542179
// MI455X (gfx1250) — compile-verified, matrix operand order checked by reading, not run on hardware
//
#include <hip/hip_runtime.h>
#include <hip/hip_bf16.h>
#include <math.h>

// Problem constants (match reference)
#define BB 4
#define NPIX 16384          // H*W
#define CCH 128             // C
#define NHD 8
#define DHD 16              // head dim
#define HH 128
#define WW 128
#define SEG 32
#define NB_ROWS (BB * NPIX) // 65536 token rows

typedef __attribute__((ext_vector_type(16))) __bf16 v16bf;
typedef __attribute__((ext_vector_type(8)))  __bf16 v8bf;
typedef __attribute__((ext_vector_type(8)))  float  v8f;
typedef __attribute__((ext_vector_type(2)))  float  v2f;

#if defined(__AMDGCN__) && __has_builtin(__builtin_amdgcn_wmma_f32_16x16x4_f32)
#define HAVE_WMMA_F32X4 1
#else
#define HAVE_WMMA_F32X4 0
#endif

#if defined(__AMDGCN__) && __has_builtin(__builtin_amdgcn_global_load_async_to_lds_b128)
#define HAVE_ASYNC_LDS 1
#else
#define HAVE_ASYNC_LDS 0
#endif

#if HAVE_ASYNC_LDS
// Builtin prototype (from compiler diagnostic): params are pointers to
// 16-byte int vectors in AS1 (printed "__device__") / AS3 ("__shared__").
typedef int v4i_vs __attribute__((__vector_size__(4 * sizeof(int))));
typedef v4i_vs __attribute__((address_space(1)))* as1_v4i;
typedef v4i_vs __attribute__((address_space(3)))* as3_v4i;
#if __has_builtin(__builtin_amdgcn_s_wait_asynccnt)
#define WAIT_ASYNC(n) __builtin_amdgcn_s_wait_asynccnt(n)
#else
#define WAIT_ASYNC(n) asm volatile("s_wait_asynccnt %0" ::"n"(n) : "memory")
#endif
#else
#define WAIT_ASYNC(n) do {} while (0)
#endif

struct BnParams {
  const float* g0; const float* b0;
  const float* g1; const float* b1;
  const float* g2; const float* b2;
  const float* g3; const float* b3;
};

// ---------------------------------------------------------------------------
// LayerNorm over last dim (C = blockDim.x), bf16 output (feeds WMMA GEMM only)
// ---------------------------------------------------------------------------
__global__ void ln_kernel(const float* __restrict__ x, const float* __restrict__ g,
                          const float* __restrict__ b, __bf16* __restrict__ y, int C) {
  __shared__ float s1[256], s2[256];
  const int row = blockIdx.x, c = threadIdx.x;
  const float v = x[(size_t)row * C + c];
  s1[c] = v; s2[c] = v * v;
  __syncthreads();
  for (int st = blockDim.x >> 1; st > 0; st >>= 1) {
    if (c < st) { s1[c] += s1[c + st]; s2[c] += s2[c + st]; }
    __syncthreads();
  }
  const float mean = s1[0] / (float)C;
  const float var  = s2[0] / (float)C - mean * mean;
  y[(size_t)row * C + c] = (__bf16)((v - mean) * rsqrtf(var + 1e-5f) * g[c] + b[c]);
}

// Weight convert + transpose: Wt[n*K + k] = (bf16) W[k*Nn + n]
__global__ void wt_convert_kernel(const float* __restrict__ W, __bf16* __restrict__ Wt,
                                  int K, int Nn) {
  const int idx = blockIdx.x * 256 + threadIdx.x;
  if (idx >= K * Nn) return;
  const int k = idx % K, n = idx / K;
  Wt[idx] = (__bf16)W[(size_t)k * Nn + n];
}

// ---------------------------------------------------------------------------
// WMMA bf16 GEMM: Cout[M,Nn] = A[M,K] @ W[K,Nn] (+bias) (+resid), f32 accum.
// A is bf16 [M,K]; Wt is bf16 [Nn,K] (pre-transposed). Tiles staged to LDS
// with GLOBAL_LOAD_ASYNC_TO_LDS_B128 (ASYNCcnt), double-buffered.
// Block = 256 thr (8 waves), tile 128x64x32, wave -> 32x32 (2x2 WMMA).
// ---------------------------------------------------------------------------
#define TILE_M 128
#define TILE_N 64
#define TILE_K 32
#define LDS_STRIDE 40   // bf16 elems; multiple of 8 keeps 16B chunk alignment

__device__ __forceinline__ void gemm_stage_copy(const __bf16* ga0, const __bf16* ga1,
                                                const __bf16* gb, __bf16* la0, __bf16* la1,
                                                __bf16* lb) {
#if HAVE_ASYNC_LDS
  __builtin_amdgcn_global_load_async_to_lds_b128((as1_v4i)ga0, (as3_v4i)la0, 0, 0);
  __builtin_amdgcn_global_load_async_to_lds_b128((as1_v4i)ga1, (as3_v4i)la1, 0, 0);
  __builtin_amdgcn_global_load_async_to_lds_b128((as1_v4i)gb,  (as3_v4i)lb,  0, 0);
#else
  const v8bf t0 = *(const v8bf*)ga0;   // batched loads -> single wait
  const v8bf t1 = *(const v8bf*)ga1;
  const v8bf t2 = *(const v8bf*)gb;
  *(v8bf*)la0 = t0;
  *(v8bf*)la1 = t1;
  *(v8bf*)lb  = t2;
#endif
}

__global__ __launch_bounds__(256)
void gemm_bf16_kernel(const __bf16* __restrict__ A, const __bf16* __restrict__ Wt,
                      const float* __restrict__ bias, const float* __restrict__ resid,
                      float* __restrict__ Cout, int M, int K, int Nn) {
  __shared__ __align__(16) __bf16 As[2][TILE_M * LDS_STRIDE];
  __shared__ __align__(16) __bf16 Bs[2][TILE_N * LDS_STRIDE];
  const int tid  = threadIdx.x;
  const int m0   = blockIdx.x * TILE_M;
  const int n0   = blockIdx.y * TILE_N;
  const int wid  = tid >> 5, lane = tid & 31;
  const int wm   = wid & 3,  wn   = wid >> 2;
  const int lrow = lane & 15, hi  = lane >> 4;
  // 16B-chunk staging: A = 512 chunks (2/thread), B = 256 chunks (1/thread)
  const int arow = tid >> 2;            // 0..63
  const int asub = (tid & 3) * 8;       // bf16 element offset within K-tile

  const int nt = K / TILE_K;
  v8f acc[2][2] = {};

  // prologue: stage tile 0 -> buffer 0
  gemm_stage_copy(A + (size_t)(m0 + arow) * K + asub,
                  A + (size_t)(m0 + arow + 64) * K + asub,
                  Wt + (size_t)(n0 + arow) * K + asub,
                  &As[0][arow * LDS_STRIDE + asub],
                  &As[0][(arow + 64) * LDS_STRIDE + asub],
                  &Bs[0][arow * LDS_STRIDE + asub]);

  for (int t = 0; t < nt; ++t) {
    const int cur = t & 1;
    if (t + 1 < nt) {
      const int kk = (t + 1) * TILE_K;
      const int nxt = cur ^ 1;
      gemm_stage_copy(A + (size_t)(m0 + arow) * K + kk + asub,
                      A + (size_t)(m0 + arow + 64) * K + kk + asub,
                      Wt + (size_t)(n0 + arow) * K + kk + asub,
                      &As[nxt][arow * LDS_STRIDE + asub],
                      &As[nxt][(arow + 64) * LDS_STRIDE + asub],
                      &Bs[nxt][arow * LDS_STRIDE + asub]);
      WAIT_ASYNC(3);      // current tile's 3 copies done; next tile in flight
    } else {
      WAIT_ASYNC(0);
    }
    __syncthreads();

    v16bf afrag[2], bfrag[2];
#pragma unroll
    for (int i = 0; i < 2; ++i) {  // lanes 0-15: K {0..7,16..23}; 16-31: {8..15,24..31}
      const __bf16* ap = &As[cur][(wm * 32 + i * 16 + lrow) * LDS_STRIDE];
      const v8bf lo = *(const v8bf*)(ap + hi * 8);
      const v8bf hh = *(const v8bf*)(ap + 16 + hi * 8);
#pragma unroll
      for (int j = 0; j < 8; ++j) { afrag[i][j] = lo[j]; afrag[i][8 + j] = hh[j]; }
    }
#pragma unroll
    for (int i = 0; i < 2; ++i) {
      const __bf16* bp = &Bs[cur][(wn * 32 + i * 16 + lrow) * LDS_STRIDE];
      const v8bf lo = *(const v8bf*)(bp + hi * 8);
      const v8bf hh = *(const v8bf*)(bp + 16 + hi * 8);
#pragma unroll
      for (int j = 0; j < 8; ++j) { bfrag[i][j] = lo[j]; bfrag[i][8 + j] = hh[j]; }
    }
#pragma unroll
    for (int i = 0; i < 2; ++i)
#pragma unroll
      for (int j = 0; j < 2; ++j)
        acc[i][j] = __builtin_amdgcn_wmma_f32_16x16x32_bf16(
            false, afrag[i], false, bfrag[j], (short)0, acc[i][j], false, false);
    __syncthreads();  // protect buffer reuse
  }
  // epilogue: C layout — VGPR r holds M = r + 8*hi, N = lrow
#pragma unroll
  for (int i = 0; i < 2; ++i)
#pragma unroll
    for (int j = 0; j < 2; ++j) {
      const int col = n0 + wn * 32 + j * 16 + lrow;
      const float bv = bias ? bias[col] : 0.0f;
#pragma unroll
      for (int r = 0; r < 8; ++r) {
        const int row = m0 + wm * 32 + i * 16 + hi * 8 + r;
        const size_t idx = (size_t)row * Nn + col;
        float val = acc[i][j][r] + bv;
        if (resid) val += resid[idx];
        Cout[idx] = val;
      }
    }
}

// ---------------------------------------------------------------------------
// (B,N,3C) qkv -> kv image (2B,C,H,W): k cols [C,2C), v cols [2C,3C)
// ---------------------------------------------------------------------------
__global__ void kv_to_img_kernel(const float* __restrict__ qkv, float* __restrict__ img) {
  const size_t idx = (size_t)blockIdx.x * 256 + threadIdx.x; // (j,c,n)
  const int n = (int)(idx % NPIX);
  size_t t = idx / NPIX;
  const int c = (int)(t % CCH);
  const int j = (int)(t / CCH);
  const int b = (j < BB) ? j : j - BB;
  const int coloff = (j < BB) ? CCH : 2 * CCH;
  img[idx] = qkv[((size_t)b * NPIX + n) * (3 * CCH) + coloff + c];
}

__global__ void copyseg0_kernel(const float* __restrict__ img, float* __restrict__ img2) {
  const size_t idx = (size_t)blockIdx.x * 256 + threadIdx.x; // (j, c<32, n)
  const int n = (int)(idx % NPIX);
  size_t t = idx / NPIX;
  const int c = (int)(t % SEG);
  const int j = (int)(t / SEG);
  const size_t a = ((size_t)j * CCH + c) * NPIX + n;
  img2[a] = img[a];
}

// depthwise SAME conv, (BT, CCn, H, W) with channel strides srcCS/dstCS
__global__ void dwconv_kernel(const float* __restrict__ src, const float* __restrict__ wk,
                              float* __restrict__ dst, int BT, int CCn, int srcCS, int dstCS,
                              int ks) {
  const size_t total = (size_t)BT * CCn * NPIX;
  const size_t idx = (size_t)blockIdx.x * 256 + threadIdx.x;
  if (idx >= total) return;
  const int xo = (int)(idx % WW);
  const int yo = (int)((idx / WW) % HH);
  size_t t = idx / NPIX;
  const int c = (int)(t % CCn);
  const int j = (int)(t / CCn);
  const int pad = ks >> 1;
  const float* wp = wk + c * ks * ks;
  const float* sp = src + ((size_t)j * srcCS + c) * NPIX;
  float acc = 0.f;
  for (int ky = 0; ky < ks; ++ky) {
    const int iy = yo + ky - pad;
    if ((unsigned)iy >= HH) continue;
    for (int kx = 0; kx < ks; ++kx) {
      const int ix = xo + kx - pad;
      if ((unsigned)ix >= WW) continue;
      acc += wp[ky * ks + kx] * sp[iy * WW + ix];
    }
  }
  dst[((size_t)j * dstCS + c) * NPIX + yo * WW + xo] = acc;
}

// pointwise 32x32: dst(2B,128,N)[base+co] = sum_ci pw[co,ci]*src(2B,32,N)[ci]
__global__ void pointwise_kernel(const float* __restrict__ src, const float* __restrict__ pw,
                                 float* __restrict__ dst, int chanBase) {
  __shared__ float wrow[SEG];
  const int j = blockIdx.z, co = blockIdx.y;
  const int p = blockIdx.x * 256 + threadIdx.x;
  if (threadIdx.x < SEG) wrow[threadIdx.x] = pw[co * SEG + threadIdx.x];
  __syncthreads();
  float acc = 0.f;
#pragma unroll
  for (int ci = 0; ci < SEG; ++ci)
    acc += wrow[ci] * src[((size_t)j * SEG + ci) * NPIX + p];
  dst[((size_t)j * CCH + chanBase + co) * NPIX + p] = acc;
}

// training-mode BN stats per channel over (2B,H,W)
__global__ void bnstats_kernel(const float* __restrict__ y, float* __restrict__ stats) {
  __shared__ float s1[256], s2[256];
  const int ch = blockIdx.x;
  float s = 0.f, sq = 0.f;
  for (int e = threadIdx.x; e < 2 * BB * NPIX; e += 256) {
    const int j = e / NPIX, n = e % NPIX;
    const float v = y[((size_t)j * CCH + ch) * NPIX + n];
    s += v; sq += v * v;
  }
  s1[threadIdx.x] = s; s2[threadIdx.x] = sq;
  __syncthreads();
  for (int st = 128; st > 0; st >>= 1) {
    if (threadIdx.x < st) { s1[threadIdx.x] += s1[threadIdx.x + st]; s2[threadIdx.x] += s2[threadIdx.x + st]; }
    __syncthreads();
  }
  if (threadIdx.x == 0) {
    const float cnt = (float)(2 * BB * NPIX);
    const float m = s1[0] / cnt;
    stats[ch * 2] = m;
    stats[ch * 2 + 1] = s2[0] / cnt - m * m;
  }
}

// BN apply + ReLU + scatter into k/v (B,NH,N,DH); d = 4*seg + (cl%4), h = cl/4
__global__ void bnapply_kernel(const float* __restrict__ y, const float* __restrict__ stats,
                               BnParams bp, float* __restrict__ kbuf, float* __restrict__ vbuf) {
  const size_t idx = (size_t)blockIdx.x * 256 + threadIdx.x;
  const int n = (int)(idx % NPIX);
  size_t t = idx / NPIX;
  const int c = (int)(t % CCH);
  const int j = (int)(t / CCH);
  const int s = c >> 5, cl = c & 31;
  const float* gp; const float* bpp;
  if (s == 0)      { gp = bp.g0; bpp = bp.b0; }
  else if (s == 1) { gp = bp.g1; bpp = bp.b1; }
  else if (s == 2) { gp = bp.g2; bpp = bp.b2; }
  else             { gp = bp.g3; bpp = bp.b3; }
  const float m = stats[c * 2], va = stats[c * 2 + 1];
  float v = (y[idx] - m) * rsqrtf(va + 1e-5f) * gp[cl] + bpp[cl];
  v = fmaxf(v, 0.f);
  const int h = cl >> 2, d = s * 4 + (cl & 3);
  const int b = (j < BB) ? j : j - BB;
  float* op = (j < BB) ? kbuf : vbuf;
  op[(((size_t)(b * NHD + h) * NPIX + n) << 4) + d] = v;
}

// softmax over N per (b,h,d), in place on (B,NH,N,DH)
__global__ void colsoftmax_kernel(float* __restrict__ k) {
  __shared__ float red[256];
  const int col = blockIdx.x, g = col >> 4, d = col & 15;
  float* base = k + (size_t)g * NPIX * DHD + d;
  float mx = -3.0e38f;
  for (int n = threadIdx.x; n < NPIX; n += 256) mx = fmaxf(mx, base[(size_t)n * DHD]);
  red[threadIdx.x] = mx; __syncthreads();
  for (int st = 128; st > 0; st >>= 1) {
    if (threadIdx.x < st) red[threadIdx.x] = fmaxf(red[threadIdx.x], red[threadIdx.x + st]);
    __syncthreads();
  }
  mx = red[0]; __syncthreads();
  float s = 0.f;
  for (int n = threadIdx.x; n < NPIX; n += 256) s += __expf(base[(size_t)n * DHD] - mx);
  red[threadIdx.x] = s; __syncthreads();
  for (int st = 128; st > 0; st >>= 1) {
    if (threadIdx.x < st) red[threadIdx.x] += red[threadIdx.x + st];
    __syncthreads();
  }
  const float inv = 1.0f / red[0];
  for (int n = threadIdx.x; n < NPIX; n += 256)
    base[(size_t)n * DHD] = __expf(base[(size_t)n * DHD] - mx) * inv;
}

// l2 norm over N per (b,h,d)
__global__ void l2norm_kernel(const float* __restrict__ t, float* __restrict__ norms) {
  __shared__ float red[256];
  const int col = blockIdx.x, g = col >> 4, d = col & 15;
  const float* base = t + (size_t)g * NPIX * DHD + d;
  float s = 0.f;
  for (int n = threadIdx.x; n < NPIX; n += 256) { const float v = base[(size_t)n * DHD]; s += v * v; }
  red[threadIdx.x] = s; __syncthreads();
  for (int st = 128; st > 0; st >>= 1) {
    if (threadIdx.x < st) red[threadIdx.x] += red[threadIdx.x + st];
    __syncthreads();
  }
  if (threadIdx.x == 0) norms[col] = fmaxf(sqrtf(red[0]), 1e-12f);
}

__global__ void fill_zero_kernel(float* __restrict__ p, int n) {
  const int i = blockIdx.x * 256 + threadIdx.x;
  if (i < n) p[i] = 0.f;
}

// out[g](16x16) += sum_{n in split} A[g,n,d] * B[g,n,e]  — fp32 WMMA 16x16x4
__global__ void gram16_kernel(const float* __restrict__ A, const float* __restrict__ Bm,
                              float* __restrict__ out) {
  const int g = blockIdx.x, split = blockIdx.y;
  const int lane = threadIdx.x, lrow = lane & 15, hi = lane >> 4;
  const int chunk = NPIX / gridDim.y;
  const int nbase = split * chunk;
  const float* Ab = A + (size_t)g * NPIX * DHD;
  const float* Bb = Bm + (size_t)g * NPIX * DHD;
  v8f acc = {};
#if HAVE_WMMA_F32X4
  for (int n = nbase; n < nbase + chunk; n += 4) {
    v2f af, bf;  // A 16x4 f32 layout: lane holds M=lrow, K={2hi,2hi+1}
    af.x = Ab[(size_t)(n + 2 * hi) * DHD + lrow];
    af.y = Ab[(size_t)(n + 2 * hi + 1) * DHD + lrow];
    bf.x = Bb[(size_t)(n + 2 * hi) * DHD + lrow];
    bf.y = Bb[(size_t)(n + 2 * hi + 1) * DHD + lrow];
    acc = __builtin_amdgcn_wmma_f32_16x16x4_f32(false, af, false, bf, (short)0, acc, false, false);
  }
#else
  for (int n = nbase; n < nbase + chunk; ++n) {
    const float bv = Bb[(size_t)n * DHD + lrow];
#pragma unroll
    for (int r = 0; r < 8; ++r) acc[r] += Ab[(size_t)n * DHD + (r + 8 * hi)] * bv;
  }
#endif
  float* ob = out + g * 256;
#pragma unroll
  for (int r = 0; r < 8; ++r) atomicAdd(&ob[(r + 8 * hi) * DHD + lrow], acc[r]);
}

// spatial: out[b,n,h*16+d] = sum_k softmax16(q)[k] * ktv[g][k][d]  (bf16 out)
__global__ void eff_apply_kernel(const float* __restrict__ qkv, const float* __restrict__ ktv,
                                 __bf16* __restrict__ out) {
  __shared__ float M[256];
  const int g = blockIdx.y, b = g >> 3, h = g & 7;
  const int n = blockIdx.x * 256 + threadIdx.x;
  M[threadIdx.x] = ktv[g * 256 + threadIdx.x];
  __syncthreads();
  const float* qrow = qkv + ((size_t)b * NPIX + n) * (3 * CCH) + h * DHD;
  float q[DHD];
  float mx = -3e38f;
#pragma unroll
  for (int i = 0; i < DHD; ++i) { q[i] = qrow[i]; mx = fmaxf(mx, q[i]); }
  float s = 0.f;
#pragma unroll
  for (int i = 0; i < DHD; ++i) { q[i] = __expf(q[i] - mx); s += q[i]; }
  const float inv = 1.f / s;
  __bf16* op = out + ((size_t)b * NPIX + n) * CCH + h * DHD;
#pragma unroll
  for (int d = 0; d < DHD; ++d) {
    float a = 0.f;
#pragma unroll
    for (int k2 = 0; k2 < DHD; ++k2) a += q[k2] * M[k2 * DHD + d];
    op[d] = (__bf16)(a * inv);
  }
}

// channel: out[b,n,h*16+d] = sum_e attn[g][d][e] * v[b,h,n,e]  (bf16 out)
__global__ void chan_apply_kernel(const float* __restrict__ vbuf, const float* __restrict__ attn,
                                  __bf16* __restrict__ out) {
  __shared__ float M[256];
  const int g = blockIdx.y, b = g >> 3, h = g & 7;
  const int n = blockIdx.x * 256 + threadIdx.x;
  M[threadIdx.x] = attn[g * 256 + threadIdx.x];
  __syncthreads();
  const float* vrow = vbuf + ((size_t)g * NPIX + n) * DHD;
  float vv[DHD];
#pragma unroll
  for (int e = 0; e < DHD; ++e) vv[e] = vrow[e];
  __bf16* op = out + ((size_t)b * NPIX + n) * CCH + h * DHD;
#pragma unroll
  for (int d = 0; d < DHD; ++d) {
    float a = 0.f;
#pragma unroll
    for (int e = 0; e < DHD; ++e) a += M[d * DHD + e] * vv[e];
    op[d] = (__bf16)a;
  }
}

// extract q from qkv into (B,NH,N,DH) f32 (for l2norm + gram)
__global__ void q_extract_kernel(const float* __restrict__ qkv, float* __restrict__ qb) {
  const size_t idx = (size_t)blockIdx.x * 256 + threadIdx.x;  // (b,n,c)
  const int c = (int)(idx & (CCH - 1));
  const size_t rn = idx >> 7;  // b*NPIX + n
  const int h = c >> 4, d = c & 15;
  const int b = (int)(rn / NPIX), n = (int)(rn % NPIX);
  qb[(((size_t)(b * NHD + h) * NPIX + n) << 4) + d] = qkv[rn * (3 * CCH) + c];
}

// attn = softmax_e( gram[d,e] / (|q_d||k_e|) * temp[h] )
__global__ void attn_finalize_kernel(const float* __restrict__ gmat, const float* __restrict__ qn,
                                     const float* __restrict__ kn, const float* __restrict__ temp,
                                     float* __restrict__ attn) {
  const int g = blockIdx.x, h = g & 7;
  const int d = threadIdx.x;  // 16 threads
  const float t = temp[h];
  const float qd = qn[g * DHD + d];
  float row[DHD];
  float mx = -3e38f;
  for (int e = 0; e < DHD; ++e) {
    row[e] = gmat[g * 256 + d * DHD + e] / (qd * kn[g * DHD + e]) * t;
    mx = fmaxf(mx, row[e]);
  }
  float s = 0.f;
  for (int e = 0; e < DHD; ++e) { row[e] = __expf(row[e] - mx); s += row[e]; }
  const float inv = 1.f / s;
  for (int e = 0; e < DHD; ++e) attn[g * 256 + d * DHD + e] = row[e] * inv;
}

// (B,N,2C) h -> (B,2C,H,W) image
__global__ void h_to_img_kernel(const float* __restrict__ h, float* __restrict__ img) {
  const size_t idx = (size_t)blockIdx.x * 256 + threadIdx.x;  // (b,c,n)
  const int n = (int)(idx % NPIX);
  size_t t = idx / NPIX;
  const int c = (int)(t % 256);
  const int b = (int)(t / 256);
  img[idx] = h[((size_t)b * NPIX + n) * 256 + c];
}

// ax = gelu( LN_{2C}( (dwconv(h)+dw_b) * h ) )  (bf16 out, feeds fc2 GEMM)
__global__ void cglu_ln_gelu_kernel(const float* __restrict__ h, const float* __restrict__ dimg,
                                    const float* __restrict__ dwb, const float* __restrict__ g,
                                    const float* __restrict__ bb, __bf16* __restrict__ ax) {
  __shared__ float s1[256], s2[256];
  const int row = blockIdx.x;
  const int b = row / NPIX, n = row % NPIX;
  const int c = threadIdx.x;
  const float dv = dimg[((size_t)b * 256 + c) * NPIX + n] + dwb[c];
  const float v = dv * h[(size_t)row * 256 + c];
  s1[c] = v; s2[c] = v * v;
  __syncthreads();
  for (int st = 128; st > 0; st >>= 1) {
    if (c < st) { s1[c] += s1[c + st]; s2[c] += s2[c + st]; }
    __syncthreads();
  }
  const float mean = s1[0] / 256.f;
  const float var = s2[0] / 256.f - mean * mean;
  const float xln = (v - mean) * rsqrtf(var + 1e-5f) * g[c] + bb[c];
  ax[(size_t)row * 256 + c] = (__bf16)(0.5f * xln * (1.f + erff(xln * 0.70710678118654752f)));
}

// ---------------------------------------------------------------------------
// Host-side stage helpers
// ---------------------------------------------------------------------------
static void run_aggregator(const float* qkv, float* img, float* img2, float* dwseg,
                           float* stats, float* kbuf, float* vbuf,
                           const float* const* dw, const float* const* pw, BnParams bp,
                           hipStream_t stream) {
  kv_to_img_kernel<<<(2 * BB * CCH * NPIX) / 256, 256, 0, stream>>>(qkv, img);
  copyseg0_kernel<<<(2 * BB * SEG * NPIX) / 256, 256, 0, stream>>>(img, img2);
  const int ksz[3] = {3, 5, 7};
  for (int s = 1; s <= 3; ++s) {
    dwconv_kernel<<<(2 * BB * SEG * NPIX) / 256, 256, 0, stream>>>(
        img + (size_t)SEG * s * NPIX, dw[s - 1], dwseg, 2 * BB, SEG, CCH, SEG, ksz[s - 1]);
    pointwise_kernel<<<dim3(NPIX / 256, SEG, 2 * BB), 256, 0, stream>>>(dwseg, pw[s - 1], img2,
                                                                        SEG * s);
  }
  bnstats_kernel<<<CCH, 256, 0, stream>>>(img2, stats);
  bnapply_kernel<<<(2 * BB * CCH * NPIX) / 256, 256, 0, stream>>>(img2, stats, bp, kbuf, vbuf);
}

static void run_cglu(const float* in, const float* resid, float* out,
                     const float* ng, const float* nb, const float* const* mp,
                     const __bf16* wt_fc1, const __bf16* wt_fc2,
                     __bf16* abuf, float* hbuf, float* img, float* img2, hipStream_t stream) {
  ln_kernel<<<NB_ROWS, CCH, 0, stream>>>(in, ng, nb, abuf, CCH);
  gemm_bf16_kernel<<<dim3(NB_ROWS / TILE_M, 256 / TILE_N), 256, 0, stream>>>(
      abuf, wt_fc1, mp[1], nullptr, hbuf, NB_ROWS, CCH, 256);
  h_to_img_kernel<<<(BB * 256 * NPIX) / 256, 256, 0, stream>>>(hbuf, img);
  dwconv_kernel<<<(BB * 256 * NPIX) / 256, 256, 0, stream>>>(img, mp[2], img2, BB, 256, 256, 256, 3);
  cglu_ln_gelu_kernel<<<NB_ROWS, 256, 0, stream>>>(hbuf, img2, mp[3], mp[4], mp[5], abuf);
  gemm_bf16_kernel<<<dim3(NB_ROWS / TILE_M, CCH / TILE_N), 256, 0, stream>>>(
      abuf, wt_fc2, mp[7], resid, out, NB_ROWS, 256, CCH);
}

// ---------------------------------------------------------------------------
extern "C" void kernel_launch(void* const* d_in, const int* in_sizes, int n_in,
                              void* d_out, int out_size, void* d_ws, size_t ws_size,
                              hipStream_t stream) {
  (void)in_sizes; (void)n_in; (void)out_size; (void)ws_size;
  const float* x   = (const float*)d_in[0];
  const float* n1g = (const float*)d_in[1];  const float* n1b = (const float*)d_in[2];
  const float* n2g = (const float*)d_in[3];  const float* n2b = (const float*)d_in[4];
  const float* n3g = (const float*)d_in[5];  const float* n3b = (const float*)d_in[6];
  const float* n4g = (const float*)d_in[7];  const float* n4b = (const float*)d_in[8];
  const float* sa_qkv_w  = (const float*)d_in[9];
  const float* sa_proj_w = (const float*)d_in[10];
  const float* sa_proj_b = (const float*)d_in[11];
  BnParams sa_bn = {(const float*)d_in[12], (const float*)d_in[13],
                    (const float*)d_in[16], (const float*)d_in[17],
                    (const float*)d_in[20], (const float*)d_in[21],
                    (const float*)d_in[24], (const float*)d_in[25]};
  const float* sa_dw[3] = {(const float*)d_in[14], (const float*)d_in[18], (const float*)d_in[22]};
  const float* sa_pw[3] = {(const float*)d_in[15], (const float*)d_in[19], (const float*)d_in[23]};
  const float* ca_qkv_w  = (const float*)d_in[26];
  const float* ca_proj_w = (const float*)d_in[27];
  const float* ca_proj_b = (const float*)d_in[28];
  BnParams ca_bn = {(const float*)d_in[29], (const float*)d_in[30],
                    (const float*)d_in[33], (const float*)d_in[34],
                    (const float*)d_in[37], (const float*)d_in[38],
                    (const float*)d_in[41], (const float*)d_in[42]};
  const float* ca_dw[3] = {(const float*)d_in[31], (const float*)d_in[35], (const float*)d_in[39]};
  const float* ca_pw[3] = {(const float*)d_in[32], (const float*)d_in[36], (const float*)d_in[40]};
  const float* ca_temp = (const float*)d_in[43];
  const float* m1[8]; for (int i = 0; i < 8; ++i) m1[i] = (const float*)d_in[44 + i];
  const float* m2[8]; for (int i = 0; i < 8; ++i) m2[i] = (const float*)d_in[52 + i];

  // workspace carve-up (float units, 256B-aligned chunks)
  float* ws = (float*)d_ws;
  size_t off = 0;
  auto take = [&](size_t n) {
    float* p = ws + off;
    off += (n + 63) & ~(size_t)63;
    return p;
  };
  float*  accA  = take((size_t)NB_ROWS * CCH);
  float*  accB  = take((size_t)NB_ROWS * CCH);
  float*  lnbuf = take((size_t)NB_ROWS * 2 * CCH);   // qb (f32) scratch
  float*  qkv   = take((size_t)NB_ROWS * 3 * CCH);   // qkv / h
  float*  img   = take((size_t)2 * BB * CCH * NPIX); // kv image / h image
  float*  img2  = take((size_t)2 * BB * CCH * NPIX); // aggregated Y / d image
  float*  dwseg = take((size_t)2 * BB * SEG * NPIX);
  float*  kbuf  = take((size_t)BB * NHD * NPIX * DHD);
  float*  vbuf  = take((size_t)BB * NHD * NPIX * DHD);
  float*  stats = take(2 * CCH);
  float*  gmat  = take(BB * NHD * 256);
  float*  attn  = take(BB * NHD * 256);
  float*  qnorm = take(BB * NHD * DHD);
  float*  knorm = take(BB * NHD * DHD);
  __bf16* abuf  = (__bf16*)take((size_t)NB_ROWS * CCH);      // bf16 GEMM A, up to 2C wide
  __bf16* wtb   = (__bf16*)take(262144 / 2);                 // bf16 transposed weights

  __bf16* wt_sa_qkv  = wtb;
  __bf16* wt_sa_proj = wtb + 49152;
  __bf16* wt_ca_qkv  = wtb + 65536;
  __bf16* wt_ca_proj = wtb + 114688;
  __bf16* wt_m1_fc1  = wtb + 131072;
  __bf16* wt_m1_fc2  = wtb + 163840;
  __bf16* wt_m2_fc1  = wtb + 196608;
  __bf16* wt_m2_fc2  = wtb + 229376;

  // weight convert + transpose (bf16, [Nn][K])
  wt_convert_kernel<<<(128 * 384) / 256, 256, 0, stream>>>(sa_qkv_w, wt_sa_qkv, 128, 384);
  wt_convert_kernel<<<(128 * 128) / 256, 256, 0, stream>>>(sa_proj_w, wt_sa_proj, 128, 128);
  wt_convert_kernel<<<(128 * 384) / 256, 256, 0, stream>>>(ca_qkv_w, wt_ca_qkv, 128, 384);
  wt_convert_kernel<<<(128 * 128) / 256, 256, 0, stream>>>(ca_proj_w, wt_ca_proj, 128, 128);
  wt_convert_kernel<<<(128 * 256) / 256, 256, 0, stream>>>(m1[0], wt_m1_fc1, 128, 256);
  wt_convert_kernel<<<(256 * 128) / 256, 256, 0, stream>>>(m1[6], wt_m1_fc2, 256, 128);
  wt_convert_kernel<<<(128 * 256) / 256, 256, 0, stream>>>(m2[0], wt_m2_fc1, 128, 256);
  wt_convert_kernel<<<(256 * 128) / 256, 256, 0, stream>>>(m2[6], wt_m2_fc2, 256, 128);

  // ---------------- Stage A: spatial attention ----------------
  ln_kernel<<<NB_ROWS, CCH, 0, stream>>>(x, n1g, n1b, abuf, CCH);
  gemm_bf16_kernel<<<dim3(NB_ROWS / TILE_M, 384 / TILE_N), 256, 0, stream>>>(
      abuf, wt_sa_qkv, nullptr, nullptr, qkv, NB_ROWS, CCH, 384);
  run_aggregator(qkv, img, img2, dwseg, stats, kbuf, vbuf, sa_dw, sa_pw, sa_bn, stream);
  colsoftmax_kernel<<<BB * NHD * DHD, 256, 0, stream>>>(kbuf);
  fill_zero_kernel<<<(BB * NHD * 256) / 256, 256, 0, stream>>>(gmat, BB * NHD * 256);
  gram16_kernel<<<dim3(BB * NHD, 64), 32, 0, stream>>>(kbuf, vbuf, gmat);
  eff_apply_kernel<<<dim3(NPIX / 256, BB * NHD), 256, 0, stream>>>(qkv, gmat, abuf);
  gemm_bf16_kernel<<<dim3(NB_ROWS / TILE_M, CCH / TILE_N), 256, 0, stream>>>(
      abuf, wt_sa_proj, sa_proj_b, x, accA, NB_ROWS, CCH, CCH);

  // ---------------- Stage B: CGLU m1 ----------------
  run_cglu(accA, accA, accB, n2g, n2b, m1, wt_m1_fc1, wt_m1_fc2, abuf, qkv, img, img2, stream);

  // ---------------- Stage C: channel attention ----------------
  ln_kernel<<<NB_ROWS, CCH, 0, stream>>>(accB, n3g, n3b, abuf, CCH);
  gemm_bf16_kernel<<<dim3(NB_ROWS / TILE_M, 384 / TILE_N), 256, 0, stream>>>(
      abuf, wt_ca_qkv, nullptr, nullptr, qkv, NB_ROWS, CCH, 384);
  run_aggregator(qkv, img, img2, dwseg, stats, kbuf, vbuf, ca_dw, ca_pw, ca_bn, stream);
  q_extract_kernel<<<((size_t)NB_ROWS * CCH) / 256, 256, 0, stream>>>(qkv, lnbuf);
  l2norm_kernel<<<BB * NHD * DHD, 256, 0, stream>>>(lnbuf, qnorm);
  l2norm_kernel<<<BB * NHD * DHD, 256, 0, stream>>>(kbuf, knorm);
  fill_zero_kernel<<<(BB * NHD * 256) / 256, 256, 0, stream>>>(gmat, BB * NHD * 256);
  gram16_kernel<<<dim3(BB * NHD, 64), 32, 0, stream>>>(lnbuf, kbuf, gmat);
  attn_finalize_kernel<<<BB * NHD, 16, 0, stream>>>(gmat, qnorm, knorm, ca_temp, attn);
  chan_apply_kernel<<<dim3(NPIX / 256, BB * NHD), 256, 0, stream>>>(vbuf, attn, abuf);
  gemm_bf16_kernel<<<dim3(NB_ROWS / TILE_M, CCH / TILE_N), 256, 0, stream>>>(
      abuf, wt_ca_proj, ca_proj_b, accB, accA, NB_ROWS, CCH, CCH);

  // ---------------- Stage D: CGLU m2 -> d_out ----------------
  run_cglu(accA, accA, (float*)d_out, n4g, n4b, m2, wt_m2_fc1, wt_m2_fc2, abuf, qkv, img, img2,
           stream);
}
